// RWKVTimeMix_50010599194889
// MI455X (gfx1250) — compile-verified, matrix operand order checked by reading, not run on hardware
//
#include <hip/hip_runtime.h>
#include <hip/hip_bf16.h>

// Problem constants (reference: B=4, T=4096, D=1024)
#define BB   4
#define TT   4096
#define DD   1024
#define MTOT (BB * TT)      // 16384 rows for all GEMMs
#define NCH  32             // scan chunks
#define LCH  (TT / NCH)     // 128 steps per chunk

// GEMM tiling
#define BM   128
#define BN   128
#define BK   32
#define LDT  48             // padded LDS row stride (elements); 96B rows keep 32B alignment

typedef __attribute__((ext_vector_type(16))) __bf16 v16bf;
typedef __attribute__((ext_vector_type(8)))  float  v8f;

__device__ __forceinline__ float sigmoidf_(float x) { return 1.0f / (1.0f + __expf(-x)); }

// gfx1250 async global->LDS copy, 16B per lane (512B per wave-instruction).
// Tracked on ASYNCcnt; no VGPR staging. LDS byte offset = low 32 bits of the
// generic pointer.
__device__ __forceinline__ void async_copy_b128(const __bf16* gsrc, __bf16* lds_dst) {
    unsigned ldsa = (unsigned)(size_t)lds_dst;
    asm volatile("global_load_async_to_lds_b128 %0, %1, off"
                 :: "v"(ldsa), "v"(gsrc) : "memory");
}
__device__ __forceinline__ void wait_asynccnt0() {
    asm volatile("s_wait_asynccnt 0" ::: "memory");
}

// ---------------------------------------------------------------------------
// 1) token shift + time-mix, fp32 -> bf16 (GEMM A-matrices for k/v/r)
// ---------------------------------------------------------------------------
__global__ void prep_mix_kernel(const float* __restrict__ x,
                                const float* __restrict__ tmk,
                                const float* __restrict__ tmv,
                                const float* __restrict__ tmr,
                                __bf16* __restrict__ xk,
                                __bf16* __restrict__ xv,
                                __bf16* __restrict__ xr) {
    int i = blockIdx.x * blockDim.x + threadIdx.x;   // [0, MTOT*DD)
    int d = i & (DD - 1);
    int t = (i >> 10) & (TT - 1);
    float xc = x[i];
    float xp = (t == 0) ? 0.0f : x[i - DD];
    float mk = tmk[d], mv = tmv[d], mr = tmr[d];
    xk[i] = (__bf16)(xc * mk + xp * (1.0f - mk));
    xv[i] = (__bf16)(xc * mv + xp * (1.0f - mv));
    xr[i] = (__bf16)(xc * mr + xp * (1.0f - mr));
}

// ---------------------------------------------------------------------------
// 2) weight conversion fp32 -> bf16 (layout preserved: [N,K] row-major)
// ---------------------------------------------------------------------------
__global__ void cvt_bf16_kernel(const float* __restrict__ s,
                                __bf16* __restrict__ dst, int n) {
    int i = blockIdx.x * blockDim.x + threadIdx.x;
    if (i < n) dst[i] = (__bf16)s[i];
}

// ---------------------------------------------------------------------------
// 3) bf16 WMMA GEMM:  C[M,N] = A[M,K] * W[N,K]^T   (torch Linear, no bias)
//    Block tile 128x128x32, 256 threads = 8 waves (wave32), double-buffered
//    LDS filled by global_load_async_to_lds_b128 (ASYNCcnt).
//    Each wave: 2x4 grid of 16x16 tiles via v_wmma_f32_16x16x32_bf16.
//    act: 0 = none, 1 = sigmoid (for r projection)
// ---------------------------------------------------------------------------
__global__ __launch_bounds__(256)
void wmma_gemm_bf16_kernel(const __bf16* __restrict__ A,
                           const __bf16* __restrict__ W,
                           float* __restrict__ C,
                           int M, int K, int N, int act) {
    __shared__ __bf16 As[2][BM * LDT];
    __shared__ __bf16 Bs[2][BM * LDT];

    const int n0   = blockIdx.x * BN;
    const int m0   = blockIdx.y * BM;
    const int tid  = threadIdx.x;
    const int lane = tid & 31;
    const int wave = tid >> 5;
    const int wm   = wave & 3;   // 4 waves along M, 32 rows each
    const int wn   = wave >> 2;  // 2 waves along N, 64 cols each
    const int NT   = K / BK;

    // 512 16B-chunks per matrix tile; 256 threads -> 2 chunks each, moved
    // asynchronously into LDS with no VGPR staging.
    auto copy_tile = [&](int kt, int buf) {
        const int k0 = kt * BK;
#pragma unroll
        for (int it = 0; it < 2; ++it) {
            int ch  = tid + it * 256;      // 0..511
            int row = ch >> 2;             // 0..127
            int q   = ch & 3;              // 16B quad within 64B row
            async_copy_b128(A + (size_t)(m0 + row) * K + k0 + q * 8,
                            &As[buf][row * LDT + q * 8]);
            async_copy_b128(W + (size_t)(n0 + row) * K + k0 + q * 8,
                            &Bs[buf][row * LDT + q * 8]);
        }
    };

    v8f zero = {0.f, 0.f, 0.f, 0.f, 0.f, 0.f, 0.f, 0.f};
    v8f acc[2][4];
#pragma unroll
    for (int mi = 0; mi < 2; ++mi)
#pragma unroll
        for (int nj = 0; nj < 4; ++nj) acc[mi][nj] = zero;

    copy_tile(0, 0);
    wait_asynccnt0();
    __syncthreads();

    const int rlo   = lane & 15;          // row within 16-row fragment
    const int khalf = (lane >> 4) * 16;   // which 16-element K half this lane holds

    for (int kt = 0; kt < NT; ++kt) {
        const int cur = kt & 1;
        if (kt + 1 < NT) copy_tile(kt + 1, cur ^ 1);
        if (kt + 2 < NT) {
            // speculative prefetch of the tile after next (global_prefetch_b8)
            __builtin_prefetch(A + (size_t)(m0 + (tid >> 1)) * K + (kt + 2) * BK, 0, 0);
            __builtin_prefetch(W + (size_t)(n0 + (tid >> 1)) * K + (kt + 2) * BK, 0, 0);
        }

        v16bf afrag[2], bfrag[4];
#pragma unroll
        for (int mi = 0; mi < 2; ++mi)
            afrag[mi] = *(const v16bf*)(const void*)
                        (&As[cur][(wm * 32 + mi * 16 + rlo) * LDT + khalf]);
#pragma unroll
        for (int nj = 0; nj < 4; ++nj)
            bfrag[nj] = *(const v16bf*)(const void*)
                        (&Bs[cur][(wn * 64 + nj * 16 + rlo) * LDT + khalf]);

#pragma unroll
        for (int mi = 0; mi < 2; ++mi)
#pragma unroll
            for (int nj = 0; nj < 4; ++nj)
                acc[mi][nj] = __builtin_amdgcn_wmma_f32_16x16x32_bf16(
                    /*neg_a=*/false, afrag[mi], /*neg_b=*/false, bfrag[nj],
                    /*c_mod=*/(short)0, acc[mi][nj],
                    /*reuse_a=*/false, /*reuse_b=*/false);

        // each wave's async copies for the next tile must land before the
        // workgroup barrier releases anyone to read that buffer
        wait_asynccnt0();
        __syncthreads();
    }

    // Epilogue: C/D layout — VGPR r holds M=r (lanes 0-15) / M=r+8 (lanes 16-31), N=lane&15
    const int mup = (lane >> 4) << 3;
#pragma unroll
    for (int mi = 0; mi < 2; ++mi) {
#pragma unroll
        for (int nj = 0; nj < 4; ++nj) {
            int mbase = m0 + wm * 32 + mi * 16 + mup;
            int n     = n0 + wn * 64 + nj * 16 + (lane & 15);
#pragma unroll
            for (int r = 0; r < 8; ++r) {
                float vv = acc[mi][nj][r];
                if (act) vv = sigmoidf_(vv);
                C[(size_t)(mbase + r) * N + n] = vv;
            }
        }
    }
}

// ---------------------------------------------------------------------------
// 4a) WKV scan pass A: per-chunk state summary (m_c, Q_c), p0=-inf, q0=0
// ---------------------------------------------------------------------------
__global__ void wkv_chunk_state_kernel(const float* __restrict__ k,
                                       const float* __restrict__ v,
                                       const float* __restrict__ w,
                                       float* __restrict__ chP,
                                       float* __restrict__ chQ) {
    int i = blockIdx.x * blockDim.x + threadIdx.x;   // [0, BB*NCH*DD)
    int d = i & (DD - 1);
    int c = (i >> 10) & (NCH - 1);
    int b = i >> 15;
    float wd = w[d];
    float p = -1e30f, q = 0.0f;
    size_t base = ((size_t)b * TT + (size_t)c * LCH) * DD + d;
    for (int t = 0; t < LCH; ++t) {
        float kt = k[base + (size_t)t * DD];
        float vt = v[base + (size_t)t * DD];
        float pw = p + wd;
        float np = fmaxf(pw, kt);
        q = __expf(fmaxf(pw - np, -30.0f)) * q + __expf(fmaxf(kt - np, -30.0f)) * vt;
        p = np;
    }
    chP[i] = p;
    chQ[i] = q;
}

// ---------------------------------------------------------------------------
// 4b) WKV scan pass B: exclusive prefix over chunk summaries per channel
// ---------------------------------------------------------------------------
__global__ void wkv_carry_kernel(const float* __restrict__ chP,
                                 const float* __restrict__ chQ,
                                 const float* __restrict__ w,
                                 float* __restrict__ caP,
                                 float* __restrict__ caQ) {
    int i = blockIdx.x * blockDim.x + threadIdx.x;   // [0, BB*DD)
    int d = i & (DD - 1);
    int b = i >> 10;
    float shift = w[d] * (float)LCH;                 // p decays by w each step
    float p = -1e30f, q = 0.0f;
    for (int c = 0; c < NCH; ++c) {
        size_t idx = ((size_t)b * NCH + c) * DD + d;
        caP[idx] = p;
        caQ[idx] = q;
        float mc = chP[idx], Qc = chQ[idx];
        float ps = p + shift;
        float np = fmaxf(ps, mc);
        q = __expf(fmaxf(ps - np, -30.0f)) * q + __expf(fmaxf(mc - np, -30.0f)) * Qc;
        p = np;
    }
}

// ---------------------------------------------------------------------------
// 4c) WKV scan pass C: replay chunk with carry-in, fuse y = r*wkv -> bf16
// ---------------------------------------------------------------------------
__global__ void wkv_out_kernel(const float* __restrict__ k,
                               const float* __restrict__ v,
                               const float* __restrict__ r,
                               const float* __restrict__ wvec,
                               const float* __restrict__ uvec,
                               const float* __restrict__ caP,
                               const float* __restrict__ caQ,
                               __bf16* __restrict__ y) {
    int i = blockIdx.x * blockDim.x + threadIdx.x;   // [0, BB*NCH*DD)
    int d = i & (DD - 1);
    int c = (i >> 10) & (NCH - 1);
    int b = i >> 15;
    float wd = wvec[d], u = uvec[d];
    float p = caP[i], q = caQ[i];
    size_t base = ((size_t)b * TT + (size_t)c * LCH) * DD + d;
    for (int t = 0; t < LCH; ++t) {
        size_t idx = base + (size_t)t * DD;
        float kt = k[idx], vt = v[idx];
        float uk = u + kt;
        float np = fmaxf(p, uk);
        float so = __expf(fmaxf(p - np, -30.0f));
        float sc = __expf(fmaxf(uk - np, -30.0f));
        float wkv = (so * q + sc * vt) / (so + sc + 1e-9f);
        y[idx] = (__bf16)(r[idx] * wkv);             // r already sigmoided in GEMM epilogue
        float pw = p + wd;
        float np2 = fmaxf(pw, kt);
        q = __expf(fmaxf(pw - np2, -30.0f)) * q + __expf(fmaxf(kt - np2, -30.0f)) * vt;
        p = np2;
    }
}

// ---------------------------------------------------------------------------
// host launcher
// ---------------------------------------------------------------------------
extern "C" void kernel_launch(void* const* d_in, const int* in_sizes, int n_in,
                              void* d_out, int out_size, void* d_ws, size_t ws_size,
                              hipStream_t stream) {
    const float* x   = (const float*)d_in[0];
    const float* tmk = (const float*)d_in[1];
    const float* tmv = (const float*)d_in[2];
    const float* tmr = (const float*)d_in[3];
    const float* Wk  = (const float*)d_in[4];
    const float* Wv  = (const float*)d_in[5];
    const float* Wr  = (const float*)d_in[6];
    const float* Wo  = (const float*)d_in[7];
    const float* td  = (const float*)d_in[8];   // time_decay
    const float* tf  = (const float*)d_in[9];   // time_first
    float* out = (float*)d_out;

    char* ws = (char*)d_ws;
    size_t off = 0;
    auto carve = [&](size_t bytes) -> void* {
        void* p = ws + off;
        off += (bytes + 255) & ~(size_t)255;
        return p;
    };

    const size_t actB = (size_t)MTOT * DD;
    __bf16* xk  = (__bf16*)carve(actB * 2);
    __bf16* xv  = (__bf16*)carve(actB * 2);
    __bf16* xr  = (__bf16*)carve(actB * 2);
    __bf16* yb  = (__bf16*)carve(actB * 2);
    __bf16* Wkb = (__bf16*)carve((size_t)DD * DD * 2);
    __bf16* Wvb = (__bf16*)carve((size_t)DD * DD * 2);
    __bf16* Wrb = (__bf16*)carve((size_t)DD * DD * 2);
    __bf16* Wob = (__bf16*)carve((size_t)DD * DD * 2);
    float*  kf  = (float*)carve(actB * 4);
    float*  vf  = (float*)carve(actB * 4);
    float*  rf  = (float*)carve(actB * 4);
    float*  chP = (float*)carve((size_t)BB * NCH * DD * 4);
    float*  chQ = (float*)carve((size_t)BB * NCH * DD * 4);
    float*  caP = (float*)carve((size_t)BB * NCH * DD * 4);
    float*  caQ = (float*)carve((size_t)BB * NCH * DD * 4);

    // 1) token shift + mixes
    prep_mix_kernel<<<(MTOT * DD) / 256, 256, 0, stream>>>(x, tmk, tmv, tmr, xk, xv, xr);

    // 2) weight conversion
    const int wcount = DD * DD;
    cvt_bf16_kernel<<<(wcount + 255) / 256, 256, 0, stream>>>(Wk, Wkb, wcount);
    cvt_bf16_kernel<<<(wcount + 255) / 256, 256, 0, stream>>>(Wv, Wvb, wcount);
    cvt_bf16_kernel<<<(wcount + 255) / 256, 256, 0, stream>>>(Wr, Wrb, wcount);
    cvt_bf16_kernel<<<(wcount + 255) / 256, 256, 0, stream>>>(Wo, Wob, wcount);

    // 3) projections: k, v, r (sigmoid fused)
    dim3 ggrid(DD / BN, MTOT / BM);
    wmma_gemm_bf16_kernel<<<ggrid, 256, 0, stream>>>(xk, Wkb, kf, MTOT, DD, DD, 0);
    wmma_gemm_bf16_kernel<<<ggrid, 256, 0, stream>>>(xv, Wvb, vf, MTOT, DD, DD, 0);
    wmma_gemm_bf16_kernel<<<ggrid, 256, 0, stream>>>(xr, Wrb, rf, MTOT, DD, DD, 1);

    // 4) chunked WKV scan (exact 3-pass decomposition)
    wkv_chunk_state_kernel<<<(BB * NCH * DD) / 256, 256, 0, stream>>>(kf, vf, td, chP, chQ);
    wkv_carry_kernel<<<(BB * DD) / 256, 256, 0, stream>>>(chP, chQ, td, caP, caQ);
    wkv_out_kernel<<<(BB * NCH * DD) / 256, 256, 0, stream>>>(kf, vf, rf, td, tf, caP, caQ, yb);

    // 5) output projection
    wmma_gemm_bf16_kernel<<<ggrid, 256, 0, stream>>>(yb, Wob, out, MTOT, DD, DD, 0);
}
